// Transformer_Comm_80771154969226
// MI455X (gfx1250) — matrix operand order repaired and checked by reading, not compiled, not run on hardware
//
#include <hip/hip_runtime.h>

#define TB 256

typedef __bf16 bf16t;
typedef bf16t v16bf __attribute__((ext_vector_type(16)));
typedef float v8f  __attribute__((ext_vector_type(8)));

typedef unsigned int u32x4 __attribute__((ext_vector_type(4)));
typedef int i32x4 __attribute__((ext_vector_type(4)));
typedef int i32x8 __attribute__((ext_vector_type(8)));

#if defined(__has_builtin)
#if __has_builtin(__builtin_amdgcn_tensor_load_to_lds) && __has_builtin(__builtin_amdgcn_s_wait_tensorcnt)
#define TDM_OK 1
#endif
#endif

union Frag { unsigned u[8]; v16bf v; };

__device__ __forceinline__ unsigned short f2bf(float f) {
  union { float f; unsigned u; } a; a.f = f;
  unsigned r = a.u + 0x7FFFu + ((a.u >> 16) & 1u);   // round-to-nearest-even
  return (unsigned short)(r >> 16);
}

__device__ __forceinline__ float gelu_exact(float x) {
  return 0.5f * x * (1.0f + erff(x * 0.7071067811865476f));
}

__device__ __forceinline__ v8f vzero8() {
  v8f z;
  #pragma unroll
  for (int r = 0; r < 8; ++r) z[r] = 0.f;
  return z;
}

// -------- bf16 pre-transposed weight pool offsets inside d_ws (elements) --------
#define OFF_ENC 0u                     //  64x128 -> [128][64]   (8192)
#define OFF_QKV 8192u                  //  6 x [128][128]        (6*16384)
#define OFF_WP  106496u                //  2 x [128][128]
#define OFF_M1  139264u                //  2 x [128][128]
#define OFF_M2  172032u                //  2 x [128][128]
#define OFF_HW1 204800u                //  [128][128]
#define OFF_HW2 221184u                //  128x32 -> [32][128]   (4096)
#define W_TOTAL_ELEMS 225280u          //  450560 bytes

#ifdef TDM_OK
// One-shot contiguous TDM DMA: d_ws(bf16) -> LDS.  Issue from a single wave; all
// operands uniform (kernel args / constants) so the descriptor lands in SGPRs.
__device__ __forceinline__ void tdm_copy(const void* gsrc, unsigned lds_byte_off,
                                         unsigned nelem /* #bf16 elements */) {
  const unsigned long long ga = (unsigned long long)(size_t)gsrc;
  u32x4 g0;
  g0[0] = 1u;                                   // count=1, user descriptor
  g0[1] = lds_byte_off;                         // lds_addr
  g0[2] = (unsigned)ga;                         // global_addr[31:0]
  g0[3] = (unsigned)(ga >> 32) | (2u << 30);    // global_addr[56:32] | type=2 ("image")
  i32x8 g1;
  g1[0] = (int)(1u << 16);                      // workgroup_mask=0, data_size=1 (2B)
  g1[1] = (int)((nelem & 0xFFFFu) << 16);       // tensor_dim0[15:0]
  g1[2] = (int)((nelem >> 16) | (1u << 16));    // tensor_dim0[31:16] | tensor_dim1=1
  g1[3] = (int)((nelem & 0xFFFFu) << 16);       // tile_dim0 = nelem
  g1[4] = 1;                                    // tile_dim1=1, tile_dim2=0 (unused)
  g1[5] = (int)nelem;                           // tensor_dim0_stride[31:0]
  g1[6] = 0;                                    // stride[47:32]=0, dim1_stride lo=0
  g1[7] = 0;
  i32x4 z4; z4[0] = 0; z4[1] = 0; z4[2] = 0; z4[3] = 0;
#if defined(__clang_major__) && __clang_major__ >= 23
  i32x8 z8;
  #pragma unroll
  for (int i = 0; i < 8; ++i) z8[i] = 0;
  __builtin_amdgcn_tensor_load_to_lds(g0, g1, z4, z4, z8, 0);
#else
  __builtin_amdgcn_tensor_load_to_lds(g0, g1, z4, z4, 0);
#endif
  __builtin_amdgcn_s_wait_tensorcnt(0);
}
#endif

// Fallback: stage global fp32 weight W[K][N] into LDS as bf16 transposed Wt[n][k].
template <int K, int N>
__device__ __forceinline__ void stage_wT(const float* __restrict__ W,
                                         unsigned short* __restrict__ Wt, int tid) {
  for (int idx = tid; idx < K * N; idx += TB) {
    const int k = idx / N, n = idx - k * N;
    Wt[n * K + k] = f2bf(W[idx]);
  }
}

// Unified staging: prefer pre-converted bf16 (TDM DMA), else convert from fp32.
template <int K, int N>
__device__ __forceinline__ void stage_weight(const float* __restrict__ Wf,
                                             const unsigned short* __restrict__ Wb,
                                             unsigned short* __restrict__ w16,
                                             unsigned w16_lds_off, int wave, int tid) {
  if (Wb) {
#ifdef TDM_OK
    if (wave == 0) tdm_copy(Wb, w16_lds_off, (unsigned)(K * N));
    (void)tid;
#else
    (void)w16_lds_off; (void)wave;
    const unsigned* s = (const unsigned*)Wb;
    unsigned* d = (unsigned*)w16;
    for (int i = tid; i < (K * N) / 2; i += TB) d[i] = s[i];
#endif
  } else {
    stage_wT<K, N>(Wf, w16, tid);
  }
}

// C[64 x Ntot] = A(bf16 LDS, stride ASTR) @ Bt(bf16 LDS transposed [n][k], stride BSTR)
template <int KTILES, int NTPER, int ASTR, int BSTR>
__device__ __forceinline__ void wmma_gemm(const unsigned short* __restrict__ A,
                                          const unsigned short* __restrict__ Bt,
                                          float* __restrict__ C, int Cstride,
                                          int wave, int lane) {
  const int mtile = wave & 3;
  const int grp   = wave >> 2;
  const int lrow  = lane & 15;
  const int half  = lane >> 4;
  const unsigned short* Arow = A + (mtile * 16 + lrow) * ASTR;
  v8f acc[NTPER];
  #pragma unroll
  for (int j = 0; j < NTPER; ++j) acc[j] = vzero8();
  #pragma unroll
  for (int ks = 0; ks < KTILES; ++ks) {
    Frag a;
    #pragma unroll
    for (int v = 0; v < 8; ++v) {
      const int p = (v & 3) + 4 * half + 8 * (v >> 2) + ks * 16;   // A 16-bit layout pairs
      a.u[v] = *(const unsigned*)(Arow + 2 * p);
    }
    #pragma unroll
    for (int j = 0; j < NTPER; ++j) {
      const int n = (grp * NTPER + j) * 16 + lrow;
      Frag bfr;
      #pragma unroll
      for (int v = 0; v < 8; ++v) {
        const int p = (v & 3) + 4 * half + 8 * (v >> 2) + ks * 16;   // B 16-bit layout pairs
        bfr.u[v] = *(const unsigned*)(Bt + n * BSTR + 2 * p);
      }
      acc[j] = __builtin_amdgcn_wmma_f32_16x16x32_bf16(false, a.v, false, bfr.v,
                                                       (short)0, acc[j], false, false);
    }
  }
  #pragma unroll
  for (int j = 0; j < NTPER; ++j) {
    const int col = (grp * NTPER + j) * 16 + lrow;
    #pragma unroll
    for (int r = 0; r < 8; ++r) {
      const int row = mtile * 16 + r + 8 * half;                   // C layout: 8 rows/lane
      C[row * Cstride + col] = acc[j][r];
    }
  }
}

// LayerNorm over 128 features; 4 lanes per row, shfl-combined.
template <typename F>
__device__ __forceinline__ void row_ln_128(F getv, const float* __restrict__ g,
                                           const float* __restrict__ b,
                                           float* __restrict__ xx_out,
                                           unsigned short* __restrict__ a16_out, int tid) {
  const int row = tid >> 2;
  const int qq  = tid & 3;
  float vals[32];
  float s = 0.f, ss = 0.f;
  #pragma unroll
  for (int c = 0; c < 32; ++c) {
    float v = getv(row, qq * 32 + c);
    vals[c] = v; s += v; ss += v * v;
  }
  s += __shfl_xor(s, 1, 32);  ss += __shfl_xor(ss, 1, 32);
  s += __shfl_xor(s, 2, 32);  ss += __shfl_xor(ss, 2, 32);
  const float mu  = s * (1.f / 128.f);
  const float var = ss * (1.f / 128.f) - mu * mu;
  const float inv = rsqrtf(var + 1e-5f);
  #pragma unroll
  for (int c = 0; c < 32; ++c) {
    const int col = qq * 32 + c;
    const float v = (vals[c] - mu) * inv * g[col] + b[col];
    if (xx_out)  xx_out[row * 128 + col] = v;
    if (a16_out) a16_out[row * 128 + col] = f2bf(v);
  }
}

// ---------------- weight pre-conversion kernel (fp32 -> bf16, transposed) ----------------
extern "C" __global__ void __launch_bounds__(TB)
wprep_kernel(const float* __restrict__ W, unsigned short* __restrict__ Wt, int K, int N) {
  const int idx = blockIdx.x * TB + threadIdx.x;
  if (idx < K * N) {
    const int k = idx / N, n = idx - k * N;
    Wt[n * K + k] = f2bf(W[idx]);
  }
}

extern "C" __global__ void __launch_bounds__(TB, 1)
tf_comm_kernel(const float* __restrict__ x, const float* __restrict__ graph,
               const float* __restrict__ enc_ln_g, const float* __restrict__ enc_ln_b,
               const float* __restrict__ enc_w, const float* __restrict__ enc_b,
               const float* __restrict__ ln_g, const float* __restrict__ ln_b,
               const float* __restrict__ Wq, const float* __restrict__ bq,
               const float* __restrict__ Wk, const float* __restrict__ bk,
               const float* __restrict__ Wv, const float* __restrict__ bv,
               const float* __restrict__ Wp, const float* __restrict__ bp,
               const float* __restrict__ mask_logits,
               const float* __restrict__ ln1_g, const float* __restrict__ ln1_b,
               const float* __restrict__ mlp_w1, const float* __restrict__ mlp_b1,
               const float* __restrict__ mlp_w2, const float* __restrict__ mlp_b2,
               const float* __restrict__ ln2_g, const float* __restrict__ ln2_b,
               const float* __restrict__ hw1, const float* __restrict__ hb1,
               const float* __restrict__ hln_g, const float* __restrict__ hln_b,
               const float* __restrict__ hw2, const float* __restrict__ hb2,
               const unsigned short* __restrict__ wbf,   // pre-converted weights or null
               float* __restrict__ out, float* __restrict__ att_out,
               float* __restrict__ mask_out) {
  extern __shared__ unsigned char lds[];
  float*          xx   = (float*)(lds);                    // 64x128 fp32 master activations
  float*          scr  = (float*)(lds + 32768);            // 64x128 fp32 GEMM scratch
  unsigned short* a16  = (unsigned short*)(lds + 65536);   // 64x128 bf16 A-matrix mirror
  unsigned short* w16  = (unsigned short*)(lds + 81920);   // 128x128 bf16 weight (transposed)
  unsigned short* q16  = (unsigned short*)(lds + 114688);  // 64x128 bf16 q (feature-packed)
  unsigned short* k16  = (unsigned short*)(lds + 131072);  // 64x128 bf16 k (feature-packed)
  unsigned short* vt16 = (unsigned short*)(lds + 147456);  // 128x64 bf16 v (seq-transposed)
  unsigned short* att16 = w16;                             // aliased: 2 x [64x64] per head-pair
  const unsigned w16_lds_off = (unsigned)(size_t)(const void*)w16;  // LDS byte offset for TDM

  const int b    = blockIdx.x;
  const int tid  = threadIdx.x;
  const int wave = tid >> 5;
  const int lane = tid & 31;
  const int lrow = lane & 15;
  const int half = lane >> 4;
  const int mtile = wave & 3;
  const int grp   = wave >> 2;

  // ---------------- encoder: LN(x) -> a16 (stride 64) ----------------
  {
    const int row = tid >> 2, qq = tid & 3;
    const float* xr = x + (size_t)b * 4096 + row * 64;
    float vals[16]; float s = 0.f, ss = 0.f;
    #pragma unroll
    for (int c = 0; c < 16; ++c) { float v = xr[qq * 16 + c]; vals[c] = v; s += v; ss += v * v; }
    s += __shfl_xor(s, 1, 32); ss += __shfl_xor(ss, 1, 32);
    s += __shfl_xor(s, 2, 32); ss += __shfl_xor(ss, 2, 32);
    const float mu = s * (1.f / 64.f), var = ss * (1.f / 64.f) - mu * mu;
    const float inv = rsqrtf(var + 1e-5f);
    #pragma unroll
    for (int c = 0; c < 16; ++c) {
      const int col = qq * 16 + c;
      a16[row * 64 + col] = f2bf((vals[c] - mu) * inv * enc_ln_g[col] + enc_ln_b[col]);
    }
  }
  stage_weight<64, 128>(enc_w, wbf ? wbf + OFF_ENC : nullptr, w16, w16_lds_off, wave, tid);
  __syncthreads();
  wmma_gemm<2, 4, 64, 64>(a16, w16, scr, 128, wave, lane);
  __syncthreads();
  for (int idx = tid; idx < 8192; idx += TB) scr[idx] = gelu_exact(scr[idx] + enc_b[idx & 127]);
  __syncthreads();
  row_ln_128([&](int r, int c) { return scr[r * 128 + c]; }, ln_g, ln_b, xx, a16, tid);
  __syncthreads();

  // ---------------- transformer layers ----------------
  for (int li = 0; li < 2; ++li) {
    // q, k, v projections
    for (int t = 0; t < 3; ++t) {
      const float* W  = (t == 0 ? Wq : t == 1 ? Wk : Wv) + li * 16384;
      const float* bb = (t == 0 ? bq : t == 1 ? bk : bv) + li * 128;
      stage_weight<128, 128>(W, wbf ? wbf + OFF_QKV + (li * 3 + t) * 16384 : nullptr,
                             w16, w16_lds_off, wave, tid);
      __syncthreads();
      wmma_gemm<4, 4, 128, 128>(a16, w16, scr, 128, wave, lane);
      __syncthreads();
      for (int idx = tid; idx < 8192; idx += TB) {
        const int r = idx >> 7, c = idx & 127;
        const unsigned short h16 = f2bf(scr[idx] + bb[c]);
        if (t == 0) q16[idx] = h16;
        else if (t == 1) k16[idx] = h16;
        else vt16[c * 64 + r] = h16;   // transpose v: [feat][seq]
      }
      __syncthreads();
    }

    // attention: 2 passes x (2 heads per pass, one head per wave-group)
    const float scale = 0.17677669529663687f;  // 1/sqrt(32)
    for (int p = 0; p < 2; ++p) {
      const int h = p * 2 + grp;
      // scores = q @ k^T (per-head K=32 -> single WMMA per tile)
      v8f acc[4];
      #pragma unroll
      for (int j = 0; j < 4; ++j) acc[j] = vzero8();
      Frag a;
      #pragma unroll
      for (int v = 0; v < 8; ++v) {
        const int pk = (v & 3) + 4 * half + 8 * (v >> 2);
        a.u[v] = *(const unsigned*)(q16 + (mtile * 16 + lrow) * 128 + h * 32 + 2 * pk);
      }
      #pragma unroll
      for (int j = 0; j < 4; ++j) {
        Frag bfr;
        #pragma unroll
        for (int v = 0; v < 8; ++v) {
          const int pb = (v & 3) + 4 * half + 8 * (v >> 2);
          bfr.u[v] = *(const unsigned*)(k16 + (j * 16 + lrow) * 128 + h * 32 + 2 * pb);
        }
        acc[j] = __builtin_amdgcn_wmma_f32_16x16x32_bf16(false, a.v, false, bfr.v,
                                                         (short)0, acc[j], false, false);
      }
      // mask fusion + graph gating + row softmax (16-lane shfl reductions)
      #pragma unroll
      for (int r = 0; r < 8; ++r) {
        const int row = mtile * 16 + r + 8 * half;
        float vals[4], mx = -3.0e38f;
        #pragma unroll
        for (int j = 0; j < 4; ++j) {
          const int col = j * 16 + lrow;
          const float ml = mask_logits[(((size_t)li * 4 + h) * 64 + row) * 64 + col];
          const float m  = 1.f / (1.f + expf(-ml));
          float av = acc[j][r] * scale * m + logf(m + 1e-8f);
          const float gg = (row == col) ? 1.f : graph[(size_t)b * 4096 + row * 64 + col];
          av *= gg;
          vals[j] = av; mx = fmaxf(mx, av);
        }
        #pragma unroll
        for (int m2 = 1; m2 < 16; m2 <<= 1) mx = fmaxf(mx, __shfl_xor(mx, m2, 32));
        float s = 0.f;
        #pragma unroll
        for (int j = 0; j < 4; ++j) { vals[j] = expf(vals[j] - mx); s += vals[j]; }
        #pragma unroll
        for (int m2 = 1; m2 < 16; m2 <<= 1) s += __shfl_xor(s, m2, 32);
        const float inv = 1.f / s;
        #pragma unroll
        for (int j = 0; j < 4; ++j) {
          const int col = j * 16 + lrow;
          const float av = vals[j] * inv;
          att16[grp * 4096 + row * 64 + col] = f2bf(av);
          if (li == 1) att_out[(((size_t)b * 4 + h) * 64 + row) * 64 + col] = av;
        }
      }
      __syncthreads();
      // y = att @ v  (K=64 -> 2 ksteps), write bf16 into a16 cols [h*32, h*32+32)
      v8f yac[2]; yac[0] = vzero8(); yac[1] = vzero8();
      #pragma unroll
      for (int ks = 0; ks < 2; ++ks) {
        Frag a2;
        #pragma unroll
        for (int v = 0; v < 8; ++v) {
          const int pk = (v & 3) + 4 * half + 8 * (v >> 2) + ks * 16;
          a2.u[v] = *(const unsigned*)(att16 + grp * 4096 + (mtile * 16 + lrow) * 64 + 2 * pk);
        }
        #pragma unroll
        for (int j = 0; j < 2; ++j) {
          Frag b2;
          #pragma unroll
          for (int v = 0; v < 8; ++v) {
            const int pb = (v & 3) + 4 * half + 8 * (v >> 2) + ks * 16;
            b2.u[v] = *(const unsigned*)(vt16 + (h * 32 + j * 16 + lrow) * 64 + 2 * pb);
          }
          yac[j] = __builtin_amdgcn_wmma_f32_16x16x32_bf16(false, a2.v, false, b2.v,
                                                           (short)0, yac[j], false, false);
        }
      }
      #pragma unroll
      for (int j = 0; j < 2; ++j)
        #pragma unroll
        for (int r = 0; r < 8; ++r)
          a16[(mtile * 16 + r + 8 * half) * 128 + h * 32 + j * 16 + lrow] = f2bf(yac[j][r]);
      __syncthreads();
    }

    // output projection + residual + LN1
    stage_weight<128, 128>(Wp + li * 16384, wbf ? wbf + OFF_WP + li * 16384 : nullptr,
                           w16, w16_lds_off, wave, tid);
    __syncthreads();
    wmma_gemm<4, 4, 128, 128>(a16, w16, scr, 128, wave, lane);
    __syncthreads();
    {
      const float* bpp = bp + li * 128;
      row_ln_128([&](int r, int c) { return xx[r * 128 + c] + scr[r * 128 + c] + bpp[c]; },
                 ln1_g + li * 128, ln1_b + li * 128, xx, a16, tid);
    }
    __syncthreads();

    // MLP
    stage_weight<128, 128>(mlp_w1 + li * 16384, wbf ? wbf + OFF_M1 + li * 16384 : nullptr,
                           w16, w16_lds_off, wave, tid);
    __syncthreads();
    wmma_gemm<4, 4, 128, 128>(a16, w16, scr, 128, wave, lane);
    __syncthreads();
    for (int idx = tid; idx < 8192; idx += TB)
      a16[idx] = f2bf(gelu_exact(scr[idx] + mlp_b1[li * 128 + (idx & 127)]));
    __syncthreads();
    stage_weight<128, 128>(mlp_w2 + li * 16384, wbf ? wbf + OFF_M2 + li * 16384 : nullptr,
                           w16, w16_lds_off, wave, tid);
    __syncthreads();
    wmma_gemm<4, 4, 128, 128>(a16, w16, scr, 128, wave, lane);
    __syncthreads();
    {
      const float* b2b = mlp_b2 + li * 128;
      row_ln_128([&](int r, int c) { return xx[r * 128 + c] + scr[r * 128 + c] + b2b[c]; },
                 ln2_g + li * 128, ln2_b + li * 128, xx, a16, tid);
    }
    __syncthreads();
  }

  // ---------------- head: LN(GELU(xx@hw1+hb1)) @ hw2 + hb2 ----------------
  stage_weight<128, 128>(hw1, wbf ? wbf + OFF_HW1 : nullptr, w16, w16_lds_off, wave, tid);
  __syncthreads();
  wmma_gemm<4, 4, 128, 128>(a16, w16, scr, 128, wave, lane);
  __syncthreads();
  for (int idx = tid; idx < 8192; idx += TB) scr[idx] = gelu_exact(scr[idx] + hb1[idx & 127]);
  __syncthreads();
  row_ln_128([&](int r, int c) { return scr[r * 128 + c]; }, hln_g, hln_b,
             (float*)nullptr, a16, tid);
  __syncthreads();
  stage_weight<128, 32>(hw2, wbf ? wbf + OFF_HW2 : nullptr, w16, w16_lds_off, wave, tid);
  __syncthreads();
  wmma_gemm<4, 1, 128, 128>(a16, w16, scr, 32, wave, lane);
  __syncthreads();
  for (int idx = tid; idx < 2048; idx += TB)
    out[(size_t)b * 2048 + idx] = scr[idx] + hb2[idx & 31];

  // third output: mask_logits[NL-1], copied once by block 0
  if (b == 0)
    for (int idx = tid; idx < 16384; idx += TB) mask_out[idx] = mask_logits[16384 + idx];
}

extern "C" void kernel_launch(void* const* d_in, const int* in_sizes, int n_in,
                              void* d_out, int out_size, void* d_ws, size_t ws_size,
                              hipStream_t stream) {
  (void)in_sizes; (void)n_in; (void)out_size;
  const float* x        = (const float*)d_in[0];
  const float* graph    = (const float*)d_in[1];
  const float* enc_ln_g = (const float*)d_in[2];
  const float* enc_ln_b = (const float*)d_in[3];
  const float* enc_w    = (const float*)d_in[4];
  const float* enc_b    = (const float*)d_in[5];
  const float* ln_g     = (const float*)d_in[6];
  const float* ln_b     = (const float*)d_in[7];
  const float* Wq       = (const float*)d_in[8];
  const float* bq       = (const float*)d_in[9];
  const float* Wk       = (const float*)d_in[10];
  const float* bk       = (const float*)d_in[11];
  const float* Wv       = (const float*)d_in[12];
  const float* bv       = (const float*)d_in[13];
  const float* Wp       = (const float*)d_in[14];
  const float* bp       = (const float*)d_in[15];
  const float* mask_l   = (const float*)d_in[16];
  const float* ln1_g    = (const float*)d_in[17];
  const float* ln1_b    = (const float*)d_in[18];
  const float* mlp_w1   = (const float*)d_in[19];
  const float* mlp_b1   = (const float*)d_in[20];
  const float* mlp_w2   = (const float*)d_in[21];
  const float* mlp_b2   = (const float*)d_in[22];
  const float* ln2_g    = (const float*)d_in[23];
  const float* ln2_b    = (const float*)d_in[24];
  const float* hw1      = (const float*)d_in[25];
  const float* hb1      = (const float*)d_in[26];
  const float* hln_g    = (const float*)d_in[27];
  const float* hln_b    = (const float*)d_in[28];
  const float* hw2      = (const float*)d_in[29];
  const float* hb2      = (const float*)d_in[30];

  float* out      = (float*)d_out;
  float* att_out  = out + (size_t)2048 * 64 * 32;
  float* mask_out = att_out + (size_t)2048 * 4 * 64 * 64;

  // Pre-convert all weights to bf16 (transposed to the LDS B-matrix layout) in d_ws.
  unsigned short* wsb = nullptr;
  if (ws_size >= (size_t)W_TOTAL_ELEMS * 2) {
    wsb = (unsigned short*)d_ws;
    auto prep = [&](const float* W, unsigned off, int K, int N) {
      wprep_kernel<<<(K * N + TB - 1) / TB, TB, 0, stream>>>(W, wsb + off, K, N);
    };
    prep(enc_w, OFF_ENC, 64, 128);
    for (int li = 0; li < 2; ++li) {
      prep(Wq + li * 16384, OFF_QKV + (li * 3 + 0) * 16384, 128, 128);
      prep(Wk + li * 16384, OFF_QKV + (li * 3 + 1) * 16384, 128, 128);
      prep(Wv + li * 16384, OFF_QKV + (li * 3 + 2) * 16384, 128, 128);
      prep(Wp + li * 16384, OFF_WP + li * 16384, 128, 128);
      prep(mlp_w1 + li * 16384, OFF_M1 + li * 16384, 128, 128);
      prep(mlp_w2 + li * 16384, OFF_M2 + li * 16384, 128, 128);
    }
    prep(hw1, OFF_HW1, 128, 128);
    prep(hw2, OFF_HW2, 128, 32);
  }

  const int lds_bytes = 163840;  // 160 KB dynamic LDS (MI455X: 320 KB/WGP -> 2 WGs/WGP)
  (void)hipFuncSetAttribute((const void*)tf_comm_kernel,
                            hipFuncAttributeMaxDynamicSharedMemorySize, lds_bytes);
  tf_comm_kernel<<<2048, TB, lds_bytes, stream>>>(
      x, graph, enc_ln_g, enc_ln_b, enc_w, enc_b, ln_g, ln_b,
      Wq, bq, Wk, bk, Wv, bv, Wp, bp, mask_l,
      ln1_g, ln1_b, mlp_w1, mlp_b1, mlp_w2, mlp_b2, ln2_g, ln2_b,
      hw1, hb1, hln_g, hln_b, hw2, hb2, wsb,
      out, att_out, mask_out);
}
